// GraphRNN_67095979098939
// MI455X (gfx1250) — matrix operand order repaired and checked by reading, not compiled, not run on hardware
//
#include <hip/hip_runtime.h>
#include <hip/hip_bf16.h>

typedef _Float16 half_t;
typedef __attribute__((ext_vector_type(16))) _Float16 v16h;
typedef __attribute__((ext_vector_type(8)))  float    v8f;

// ---------------------------------------------------------------------------
// WMMA helpers (gfx1250, wave32). D = A(16x32,f16) * B(32x16,f16) + C(f32)
// ---------------------------------------------------------------------------
__device__ __forceinline__ v8f wmma16(v16h a, v16h b, v8f c) {
  return __builtin_amdgcn_wmma_f32_16x16x32_f16(
      /*neg_a=*/false, a, /*neg_b=*/false, b,
      /*c_mod=*/(short)0, c, /*reuse_a=*/false, /*reuse_b=*/false);
}

// A tile 16x32 (MxK) from f16 buffer, row-major with leading dim `ld`.
// Contiguous-k layout used here (matches load_b_f16): lanes 0-15 -> M=lane, K=0..15; lanes 16-31 -> K+=16.
__device__ __forceinline__ v16h load_a_f16(const half_t* base, int ld,
                                           int m0, int k0, int lane) {
  int m  = m0 + (lane & 15);
  int kh = (lane >> 4) << 4;           // 0 or 16
  v16h a{};
#pragma unroll
  for (int i = 0; i < 8; ++i) {
    int k = k0 + 2 * i + kh;
    a[2 * i]     = base[m * ld + k];
    a[2 * i + 1] = base[m * ld + k + 1];
  }
  return a;
}

// B tile 32x16 (KxN) where B(k,n) = base[n*ld + k]  (i.e. W stored (N,K) row-major,
// which is exactly "x @ W.T" with PyTorch-layout W). K is contiguous per column.
// Contiguous-k layout used here (matches load_a_f16): lanes 0-15 -> N=lane, K=2v,2v+1 ; lanes 16-31 -> K+=16.
__device__ __forceinline__ v16h load_b_f16(const half_t* base, int ld,
                                           int k0, int n0, int lane) {
  int n    = n0 + (lane & 15);
  int koff = k0 + ((lane >> 4) << 4);  // 0 or 16
  v16h b{};
#pragma unroll
  for (int i = 0; i < 8; ++i) {
    int k = koff + 2 * i;
    b[2 * i]     = base[n * ld + k];
    b[2 * i + 1] = base[n * ld + k + 1];
  }
  return b;
}

__device__ __forceinline__ float sigmoidf_(float x) {
  return 1.0f / (1.0f + expf(-x));
}

// ---------------------------------------------------------------------------
// Problem constants
// ---------------------------------------------------------------------------
#define BB 64      // batch
#define TT 128     // max nodes
#define MM 32      // bfs bandwidth
#define HH 256     // graph hidden
#define EE 128     // edge hidden
#define TS (TT-1)  // 127 graph steps
#define TBR (TS*BB) // 8128 edge-RNN rows

// ---------------------------------------------------------------------------
// f32 -> f16 conversion
// ---------------------------------------------------------------------------
__global__ void k_f2h(const float* __restrict__ src, half_t* __restrict__ dst, int n) {
  int i = blockIdx.x * blockDim.x + threadIdx.x;
  if (i < n) dst[i] = (half_t)src[i];
}

// Build x16 (8128 x 32): row = t*64+b ; x = (t==0) ? 0 : edge_sequences[b, t, :]
__global__ void k_x16(const float* __restrict__ es, half_t* __restrict__ x16) {
  int i = blockIdx.x * blockDim.x + threadIdx.x;
  if (i >= TBR * MM) return;
  int j = i & 31, row = i >> 5;
  int t = row >> 6, b = row & 63;
  float v = (t == 0) ? 0.0f : es[b * (TT * MM) + t * MM + j];
  x16[i] = (half_t)v;
}

// gi_all (8128 x 768) = x16 @ gW_ih.T + gb_ih    (K=32 -> one WMMA per tile)
__global__ __launch_bounds__(256) void k_gi(const half_t* __restrict__ x16,
                                            const half_t* __restrict__ Wih,   // (768,32) f16
                                            const float*  __restrict__ gbih,  // 768
                                            float*        __restrict__ gi_all) {
  int rt = blockIdx.x;             // 0..507 row tile
  int lane = threadIdx.x & 31, w = threadIdx.x >> 5;  // 8 waves
  v16h a = load_a_f16(x16, MM, rt * 16, 0, lane);
#pragma unroll
  for (int c = 0; c < 6; ++c) {
    int ctile = w * 6 + c;         // 0..47
    v16h b = load_b_f16(Wih, MM, 0, ctile * 16, lane);
    v8f acc{};
    acc = wmma16(a, b, acc);
    int ncol = ctile * 16 + (lane & 15);
    float bias = gbih[ncol];
#pragma unroll
    for (int e = 0; e < 8; ++e) {
      int m = rt * 16 + e + ((lane >> 4) << 3);
      gi_all[(size_t)m * (3 * HH) + ncol] = acc[e] + bias;
    }
  }
}

// ---------------------------------------------------------------------------
// Graph-level GRU: single workgroup (32 waves), 127 sequential steps.
// h kept in LDS as f16 (matmul operand); fp32 h kept in the owning wave's regs.
// Each wave owns 2 of the 64 (4x16) 16x16 h-tiles -> 3 gate GEMM tiles each.
// ---------------------------------------------------------------------------
__global__ __launch_bounds__(1024) void k_graph_gru(
    const float*  __restrict__ gi_all,  // (8128, 768)
    const half_t* __restrict__ Whh,     // (768, 256) f16
    const float*  __restrict__ gbhh,    // 768
    const float*  __restrict__ h0,      // 256
    half_t*       __restrict__ gs16) {  // (8128, 256) f16 out
  __shared__ half_t h16[BB * HH];
  const int tid = threadIdx.x, lane = tid & 31, w = tid >> 5;  // w: 0..31
  float hprev[2][8];

  for (int i = tid; i < BB * HH; i += 1024) h16[i] = (half_t)h0[i & (HH - 1)];
#pragma unroll
  for (int j = 0; j < 2; ++j) {
    int tt = 2 * w + j, ct = tt & 15;
    float hv = h0[ct * 16 + (lane & 15)];
#pragma unroll
    for (int e = 0; e < 8; ++e) hprev[j][e] = hv;
  }

  for (int t = 0; t < TS; ++t) {
    __syncthreads();                       // h16 for this step is visible
    float hnew[2][8];
#pragma unroll
    for (int j = 0; j < 2; ++j) {
      int tt = 2 * w + j, rt = tt >> 4, ct = tt & 15;
      v8f ar{}, az{}, an{};
#pragma unroll
      for (int kc = 0; kc < HH / 32; ++kc) {
        v16h a  = load_a_f16(h16, HH, rt * 16, kc * 32, lane);
        v16h br = load_b_f16(Whh, HH, kc * 32, 0 * HH + ct * 16, lane);
        v16h bz = load_b_f16(Whh, HH, kc * 32, 1 * HH + ct * 16, lane);
        v16h bn = load_b_f16(Whh, HH, kc * 32, 2 * HH + ct * 16, lane);
        ar = wmma16(a, br, ar);
        az = wmma16(a, bz, az);
        an = wmma16(a, bn, an);
      }
      int ncol = ct * 16 + (lane & 15);
      float bhr = gbhh[ncol], bhz = gbhh[HH + ncol], bhn = gbhh[2 * HH + ncol];
#pragma unroll
      for (int e = 0; e < 8; ++e) {
        int m = rt * 16 + e + ((lane >> 4) << 3);
        const float* gi = gi_all + (size_t)(t * BB + m) * (3 * HH);
        float rr = sigmoidf_(gi[ncol]          + ar[e] + bhr);
        float zz = sigmoidf_(gi[HH + ncol]     + az[e] + bhz);
        float nn = tanhf   (gi[2 * HH + ncol] + rr * (an[e] + bhn));
        hnew[j][e] = (1.0f - zz) * nn + zz * hprev[j][e];
      }
    }
    __syncthreads();                       // all reads of old h16 done
#pragma unroll
    for (int j = 0; j < 2; ++j) {
      int tt = 2 * w + j, rt = tt >> 4, ct = tt & 15;
      int ncol = ct * 16 + (lane & 15);
#pragma unroll
      for (int e = 0; e < 8; ++e) {
        int m = rt * 16 + e + ((lane >> 4) << 3);
        half_t hv = (half_t)hnew[j][e];
        h16[m * HH + ncol] = hv;
        gs16[(size_t)(t * BB + m) * HH + ncol] = hv;
        hprev[j][e] = hnew[j][e];
      }
    }
  }
}

// h_edge (8128 x 128) = g_states @ ei_W.T + ei_b
__global__ __launch_bounds__(256) void k_hedge(const half_t* __restrict__ gs16,  // (8128,256)
                                               const half_t* __restrict__ eiW,   // (128,256) f16
                                               const float*  __restrict__ eib,   // 128
                                               float*        __restrict__ hedge) {
  int rt = blockIdx.x;                       // 0..507
  int lane = threadIdx.x & 31, w = threadIdx.x >> 5;  // ct = w (0..7)
  v8f acc{};
#pragma unroll
  for (int kc = 0; kc < HH / 32; ++kc) {
    v16h a = load_a_f16(gs16, HH, rt * 16, kc * 32, lane);
    v16h b = load_b_f16(eiW, HH, kc * 32, w * 16, lane);
    acc = wmma16(a, b, acc);
  }
  int ncol = w * 16 + (lane & 15);
  float bias = eib[ncol];
#pragma unroll
  for (int e = 0; e < 8; ++e) {
    int m = rt * 16 + e + ((lane >> 4) << 3);
    hedge[(size_t)m * EE + ncol] = acc[e] + bias;
  }
}

// ---------------------------------------------------------------------------
// Edge-level GRU + output MLP + masked BCE. One block per graph step t
// (rows t*64..t*64+63 are batch-independent), 32 sequential edge steps.
// gi_e is rank-1 (input is a 0/1 bit) => pure elementwise, no GEMM needed.
// ---------------------------------------------------------------------------
__global__ __launch_bounds__(256) void k_edge_gru(
    const float*  __restrict__ hedge,   // (8128,128)
    const half_t* __restrict__ Whh,     // (384,128) f16
    const float*  __restrict__ ebhh,    // 384
    const float*  __restrict__ ebih,    // 384
    const float*  __restrict__ eWih,    // 384 (column vector (384,1))
    const half_t* __restrict__ o1W16,   // (32,128) f16
    const float*  __restrict__ o1b,     // 32
    const float*  __restrict__ o2W,     // 32
    const float*  __restrict__ o2b,     // 1
    const float*  __restrict__ es,      // (64,128,32)
    const int*    __restrict__ lengths, // 64
    float* __restrict__ pbce, float* __restrict__ pcnt) {
  const int tblk = blockIdx.x;          // 0..126, ts = tblk+1
  const int tid = threadIdx.x, lane = tid & 31, w = tid >> 5;  // 8 waves
  __shared__ half_t he16[BB * EE];
  __shared__ float  hid[BB * 32];
  __shared__ float  ibits[BB];
  __shared__ float  red[BB];
  __shared__ float  redc[BB];

  const float* hrow = hedge + (size_t)tblk * BB * EE;
  for (int i = tid; i < BB * EE; i += 256) he16[i] = (half_t)hrow[i];

  const int rt = w >> 1, ctb = (w & 1) * 4;   // 4 owned h-tiles per wave
  float hprev[4][8];
#pragma unroll
  for (int j = 0; j < 4; ++j) {
    int ncol = (ctb + j) * 16 + (lane & 15);
#pragma unroll
    for (int e = 0; e < 8; ++e) {
      int m = rt * 16 + e + ((lane >> 4) << 3);
      hprev[j][e] = hrow[m * EE + ncol];
    }
  }

  const int ts = tblk + 1;
  const int mvalid = ts < MM ? ts : MM;
  float lsum = 0.0f, lcnt = 0.0f;

  for (int step = 0; step < MM; ++step) {
    if (tid < BB)
      ibits[tid] = (step == 0) ? 0.0f : es[tid * (TT * MM) + ts * MM + (step - 1)];
    __syncthreads();                    // he16 + ibits ready

    float hnew[4][8];
#pragma unroll
    for (int j = 0; j < 4; ++j) {
      int ct = ctb + j;
      v8f ar{}, az{}, an{};
#pragma unroll
      for (int kc = 0; kc < EE / 32; ++kc) {
        v16h a  = load_a_f16(he16, EE, rt * 16, kc * 32, lane);
        v16h br = load_b_f16(Whh, EE, kc * 32, 0 * EE + ct * 16, lane);
        v16h bz = load_b_f16(Whh, EE, kc * 32, 1 * EE + ct * 16, lane);
        v16h bn = load_b_f16(Whh, EE, kc * 32, 2 * EE + ct * 16, lane);
        ar = wmma16(a, br, ar);
        az = wmma16(a, bz, az);
        an = wmma16(a, bn, an);
      }
      int ncol = ct * 16 + (lane & 15);
      float bhr = ebhh[ncol], bhz = ebhh[EE + ncol], bhn = ebhh[2 * EE + ncol];
      float bir = ebih[ncol], biz = ebih[EE + ncol], bin = ebih[2 * EE + ncol];
      float wir = eWih[ncol], wiz = eWih[EE + ncol], win = eWih[2 * EE + ncol];
#pragma unroll
      for (int e = 0; e < 8; ++e) {
        int m = rt * 16 + e + ((lane >> 4) << 3);
        float xb = ibits[m];
        float rr = sigmoidf_(bir + xb * wir + ar[e] + bhr);
        float zz = sigmoidf_(biz + xb * wiz + az[e] + bhz);
        float nn = tanhf   (bin + xb * win + rr * (an[e] + bhn));
        hnew[j][e] = (1.0f - zz) * nn + zz * hprev[j][e];
      }
    }
    __syncthreads();                    // reads of old he16 done
#pragma unroll
    for (int j = 0; j < 4; ++j) {
      int ncol = (ctb + j) * 16 + (lane & 15);
#pragma unroll
      for (int e = 0; e < 8; ++e) {
        int m = rt * 16 + e + ((lane >> 4) << 3);
        he16[m * EE + ncol] = (half_t)hnew[j][e];
        hprev[j][e] = hnew[j][e];
      }
    }
    __syncthreads();                    // new he16 visible
    {                                   // hid = relu(h @ o1W.T + o1b) : 8 tiles
      int rt2 = w & 3, ct2 = w >> 2;
      v8f acc{};
#pragma unroll
      for (int kc = 0; kc < EE / 32; ++kc) {
        v16h a = load_a_f16(he16, EE, rt2 * 16, kc * 32, lane);
        v16h b = load_b_f16(o1W16, EE, kc * 32, ct2 * 16, lane);
        acc = wmma16(a, b, acc);
      }
      int ncol = ct2 * 16 + (lane & 15);
      float bias = o1b[ncol];
#pragma unroll
      for (int e = 0; e < 8; ++e) {
        int m = rt2 * 16 + e + ((lane >> 4) << 3);
        float v = acc[e] + bias;
        hid[m * 32 + ncol] = v > 0.0f ? v : 0.0f;
      }
    }
    __syncthreads();
    if (tid < BB) {
      float l = o2b[0];
#pragma unroll
      for (int j = 0; j < 32; ++j) l += hid[tid * 32 + j] * o2W[j];
      float tgt = es[tid * (TT * MM) + ts * MM + step];
      bool mk = (step < mvalid) && (lengths[tid] > ts);
      float bce = fmaxf(l, 0.0f) - l * tgt + log1pf(expf(-fabsf(l)));
      if (mk) { lsum += bce; lcnt += 1.0f; }
    }
    __syncthreads();
  }

  if (tid < BB) { red[tid] = lsum; redc[tid] = lcnt; }
  __syncthreads();
  if (tid == 0) {
    float s = 0.0f, c = 0.0f;
    for (int i = 0; i < BB; ++i) { s += red[i]; c += redc[i]; }
    pbce[tblk] = s; pcnt[tblk] = c;
  }
}

__global__ void k_final(const float* __restrict__ pbce,
                        const float* __restrict__ pcnt,
                        float* __restrict__ out) {
  if (threadIdx.x == 0 && blockIdx.x == 0) {
    float s = 0.0f, c = 0.0f;
    for (int i = 0; i < TS; ++i) { s += pbce[i]; c += pcnt[i]; }
    out[0] = s / fmaxf(c, 1.0f);
  }
}

// ---------------------------------------------------------------------------
extern "C" void kernel_launch(void* const* d_in, const int* in_sizes, int n_in,
                              void* d_out, int out_size, void* d_ws, size_t ws_size,
                              hipStream_t stream) {
  (void)in_sizes; (void)n_in; (void)out_size; (void)ws_size;
  const float* es    = (const float*)d_in[0];
  const int*   len   = (const int*)  d_in[1];
  const float* h0    = (const float*)d_in[2];
  const float* gWih  = (const float*)d_in[3];
  const float* gWhh  = (const float*)d_in[4];
  const float* gbih  = (const float*)d_in[5];
  const float* gbhh  = (const float*)d_in[6];
  const float* eiW   = (const float*)d_in[7];
  const float* eib   = (const float*)d_in[8];
  const float* eWih  = (const float*)d_in[9];
  const float* eWhh  = (const float*)d_in[10];
  const float* ebih  = (const float*)d_in[11];
  const float* ebhh  = (const float*)d_in[12];
  const float* o1W   = (const float*)d_in[13];
  const float* o1b   = (const float*)d_in[14];
  const float* o2W   = (const float*)d_in[15];
  const float* o2b   = (const float*)d_in[16];
  float* out = (float*)d_out;

  // workspace carve-up (256B aligned)
  char* p = (char*)d_ws;
  auto carve = [&](size_t bytes) -> void* {
    void* r = (void*)p; p += (bytes + 255) & ~(size_t)255; return r;
  };
  float*  gi_all  = (float*) carve((size_t)TBR * 3 * HH * 4);
  half_t* gs16    = (half_t*)carve((size_t)TBR * HH * 2);
  float*  hedge   = (float*) carve((size_t)TBR * EE * 4);
  half_t* x16     = (half_t*)carve((size_t)TBR * MM * 2);
  half_t* gWih16  = (half_t*)carve((size_t)3 * HH * MM * 2);
  half_t* gWhh16  = (half_t*)carve((size_t)3 * HH * HH * 2);
  half_t* eiW16   = (half_t*)carve((size_t)EE * HH * 2);
  half_t* eWhh16  = (half_t*)carve((size_t)3 * EE * EE * 2);
  half_t* o1W16   = (half_t*)carve((size_t)32 * EE * 2);
  float*  pbce    = (float*) carve((size_t)TS * 4);
  float*  pcnt    = (float*) carve((size_t)TS * 4);

  auto cvt = [&](const float* s, half_t* d, int n) {
    k_f2h<<<(n + 255) / 256, 256, 0, stream>>>(s, d, n);
  };
  cvt(gWih, gWih16, 3 * HH * MM);
  cvt(gWhh, gWhh16, 3 * HH * HH);
  cvt(eiW,  eiW16,  EE * HH);
  cvt(eWhh, eWhh16, 3 * EE * EE);
  cvt(o1W,  o1W16,  32 * EE);

  k_x16<<<(TBR * MM + 255) / 256, 256, 0, stream>>>(es, x16);
  k_gi<<<TBR / 16, 256, 0, stream>>>(x16, gWih16, gbih, gi_all);
  k_graph_gru<<<1, 1024, 0, stream>>>(gi_all, gWhh16, gbhh, h0, gs16);
  k_hedge<<<TBR / 16, 256, 0, stream>>>(gs16, eiW16, eib, hedge);
  k_edge_gru<<<TS, 256, 0, stream>>>(hedge, eWhh16, ebhh, ebih, eWih,
                                     o1W16, o1b, o2W, o2b, es, len, pbce, pcnt);
  k_final<<<1, 32, 0, stream>>>(pbce, pcnt, out);
}
